// MACEInteraction_47004122087965
// MI455X (gfx1250) — compile-verified, matrix operand order checked by reading, not run on hardware
//
#include <hip/hip_runtime.h>
#include <hip/hip_bf16.h>
#include <math.h>

// ---------------------------------------------------------------------------
// MACE interaction for MI455X (gfx1250): bf16 WMMA GEMMs + L2-resident scatter
// Round 3: force native global_atomic_add_f32 via relaxed agent-scope atomics.
// ---------------------------------------------------------------------------

typedef __bf16 bf16_t;
typedef __attribute__((ext_vector_type(16))) __bf16 v16bf;
typedef __attribute__((ext_vector_type(8)))  __bf16 v8bf;
typedef __attribute__((ext_vector_type(8)))  float  v8f;

#define N_NODES 5000
#define N_EDGES 100000
#define MUL     128

__device__ __forceinline__ v8f zero_v8f() {
  v8f z;
#pragma unroll
  for (int i = 0; i < 8; ++i) z[i] = 0.0f;
  return z;
}

__device__ __forceinline__ v8f wmma_bf16(v16bf a, v16bf b, v8f c) {
  return __builtin_amdgcn_wmma_f32_16x16x32_bf16(
      false, a, false, b, (short)0, c, false, false);
}

__device__ __forceinline__ float4 ld4(const float* p) {
  return *(const float4*)p;
}

// Relaxed, agent-scope f32 atomic add -> GLOBAL_ATOMIC_ADD_F32 (no return,
// no ordering cost; segment-sum is order-insensitive).
__device__ __forceinline__ void atomic_add_relaxed(float* p, float v) {
  __hip_atomic_fetch_add(p, v, __ATOMIC_RELAXED, __HIP_MEMORY_SCOPE_AGENT);
}

// ---------------------------------------------------------------------------
// Fragment swizzle: B matrix element (k, n) of a (32*KT) x (16*NT) matrix is
// stored so that each lane's 16 fragment elements are contiguous (32 bytes).
// ISA 16-bit layout: lane = n%16 + 16*half, element j = group*8 + within,
// with k = ktile*32 + group*16 + half*8 + within.
// ---------------------------------------------------------------------------
__device__ __forceinline__ int bsw_index(int k, int n, int ntiles) {
  const int ktile = k >> 5, kk = k & 31;
  const int ntile = n >> 4, nn = n & 15;
  const int half = (kk >> 3) & 1, group = kk >> 4, within = kk & 7;
  return (((ktile * ntiles + ntile) * 32) + half * 16 + nn) * 16 + group * 8 + within;
}

// A-stage swizzle (same pattern, lane = m + 16*half), single column of tiles.
__device__ __forceinline__ int asw_index(int m, int k) {
  const int ktile = k >> 5, kk = k & 31;
  const int half = (kk >> 3) & 1, group = kk >> 4, within = kk & 7;
  return ((ktile * 32) + half * 16 + m) * 16 + group * 8 + within;
}

// One fragment = 32 contiguous bytes per lane = 2x ds_load_b128.
__device__ __forceinline__ v16bf load_frag_lds(const bf16_t* W, int tile, int lane) {
  const bf16_t* p = W + ((size_t)tile * 32 + lane) * 16;
  const v8bf lo = *(const v8bf*)p;
  const v8bf hi = *(const v8bf*)(p + 8);
  v16bf b;
#pragma unroll
  for (int j = 0; j < 8; ++j) { b[j] = lo[j]; b[j + 8] = hi[j]; }
  return b;
}

// A-fragment (16x32) from row-major f32 global memory.  Branchless: rows
// past nrows are clamped (a garbage A row only affects the same D row, which
// is never stored).  4x global_load_b128 + 8x v_cvt_pk_bf16_f32.
__device__ __forceinline__ v16bf load_a_gmem(const float* __restrict__ A, int ld,
                                             int row0, int nrows, int k0, int lane) {
  int m = row0 + (lane & 15);
  m = (m < nrows) ? m : (nrows - 1);
  const int kh = ((lane >> 4) & 1) * 8;
  const float* p0 = A + (size_t)m * ld + k0 + kh;   // K run 0: 8 floats
  const float* p1 = p0 + 16;                        // K run 1: 8 floats
  float t[16];
  *(float4*)(t + 0)  = ld4(p0);
  *(float4*)(t + 4)  = ld4(p0 + 4);
  *(float4*)(t + 8)  = ld4(p1);
  *(float4*)(t + 12) = ld4(p1 + 4);
  v16bf a;
#pragma unroll
  for (int j = 0; j < 16; ++j) a[j] = (bf16_t)t[j];
  return a;
}

__device__ __forceinline__ float silu_f(float v) {
  return v / (1.0f + __expf(-v));
}

// Write 4 C/D tiles (16x64 span) into the bf16 A-swizzled stage with SiLU.
// C/D layout: VGPR r, lanes 0-15 -> (M=r, N=lane); lanes 16-31 -> (M=8+r).
__device__ __forceinline__ void store_silu(bf16_t* stg, const v8f* h, int lane) {
  const int mh  = (lane >> 4) << 3;
  const int col = lane & 15;
#pragma unroll
  for (int nt = 0; nt < 4; ++nt) {
#pragma unroll
    for (int r = 0; r < 8; ++r) {
      stg[asw_index(r + mh, nt * 16 + col)] = (bf16_t)silu_f(h[nt][r]);
    }
  }
}

// ---------------------------------------------------------------------------
__global__ void zero_kernel(float* __restrict__ p, int n) {
  int i = blockIdx.x * blockDim.x + threadIdx.x;
  const int stride = gridDim.x * blockDim.x;
  for (; i < n; i += stride) p[i] = 0.0f;
}

// Stage B(128x128) in static LDS (32 KB, fragment-swizzled bf16) per block.
__device__ __forceinline__ void stage_B128(const float* __restrict__ B, bf16_t* Bs) {
  const int tid = threadIdx.x;
  for (int idx = tid; idx < 128 * 128; idx += 256) {
    const int k = idx >> 7, n = idx & 127;
    Bs[bsw_index(k, n, 8)] = (bf16_t)B[idx];
  }
  __syncthreads();
}

// x = node_features @ W_pre * inv. grid(ceil(N/16)), block 256 (8 waves),
// wave w owns output columns [16w, 16w+16).
__global__ void gemm_pre_kernel(const float* __restrict__ NF,
                                const float* __restrict__ Wpre,
                                float* __restrict__ X) {
  __shared__ bf16_t Bs[128 * 128];
  stage_B128(Wpre, Bs);

  const int lane = threadIdx.x & 31;
  const int wave = threadIdx.x >> 5;
  const int row0 = blockIdx.x * 16;
  v8f acc = zero_v8f();
#pragma unroll
  for (int kt = 0; kt < 4; ++kt) {
    acc = wmma_bf16(load_a_gmem(NF, 128, row0, N_NODES, kt * 32, lane),
                    load_frag_lds(Bs, kt * 8 + wave, lane), acc);
  }
  const float inv = 0.08838834764831845f;
  const int mh  = (lane >> 4) << 3;
  const int col = wave * 16 + (lane & 15);
#pragma unroll
  for (int r = 0; r < 8; ++r) {
    const int m = row0 + r + mh;
    if (m < N_NODES) X[(size_t)m * 128 + col] = acc[r] * inv;
  }
}

// out[n, v, c] = inv * sum_u tot[c][n][u] * W_post[l(c)][u][v]
__global__ void gemm_post_kernel(const float* __restrict__ tot,
                                 const float* __restrict__ W_post,
                                 float* __restrict__ out) {
  __shared__ bf16_t Bs[128 * 128];
  const int c = blockIdx.y;                        // channel 0..15
  const int l = (c == 0) ? 0 : (c < 4) ? 1 : (c < 9) ? 2 : 3;
  const float* A = tot + (size_t)c * (N_NODES * 128);
  stage_B128(W_post + (size_t)l * 128 * 128, Bs);

  const int lane = threadIdx.x & 31;
  const int wave = threadIdx.x >> 5;
  const int row0 = blockIdx.x * 16;
  v8f acc = zero_v8f();
#pragma unroll
  for (int kt = 0; kt < 4; ++kt) {
    acc = wmma_bf16(load_a_gmem(A, 128, row0, N_NODES, kt * 32, lane),
                    load_frag_lds(Bs, kt * 8 + wave, lane), acc);
  }
  const float inv = 0.08838834764831845f;
  const int mh  = (lane >> 4) << 3;
  const int col = wave * 16 + (lane & 15);
#pragma unroll
  for (int r = 0; r < 8; ++r) {
    const int m = row0 + r + mh;
    if (m < N_NODES) out[((size_t)m * 128 + col) * 16 + c] = acc[r] * inv;
  }
}

// ---------------------------------------------------------------------------
// Fused edge kernel: radial MLP (bf16 WMMA, swizzled weights in LDS with
// 1/sqrt(fan_in) folded in) + gather of x[edge_j] + scatter-add into the
// channel-major accumulator (fits in the 192 MB L2).
// ---------------------------------------------------------------------------
__global__ void edge_kernel(const float* __restrict__ rb,
                            const float* __restrict__ sph,
                            const int*   __restrict__ edge_i,
                            const int*   __restrict__ edge_j,
                            const float* __restrict__ x,
                            const float* __restrict__ w0,
                            const float* __restrict__ w1,
                            const float* __restrict__ w2,
                            const float* __restrict__ w3,
                            float* __restrict__ tot) {
  extern __shared__ char smem_raw[];
  bf16_t* W0s   = (bf16_t*)smem_raw;        // 32x64  (K padded 8->32), NT=4
  bf16_t* W1s   = W0s + 32 * 64;            // 64x64, NT=4
  bf16_t* W2s   = W1s + 64 * 64;            // 64x64, NT=4
  bf16_t* W3s   = W2s + 64 * 64;            // 64x512, NT=32
  bf16_t* stage = W3s + 64 * 512;           // 8 waves x (2 ktiles x 512) bf16
  float*  sphs  = (float*)(stage + 8 * 1024);  // 8 waves x (16x16) f32
  int*    eis   = (int*)(sphs + 8 * 256);      // 128
  int*    ejs   = eis + 128;                   // 128

  const int tid  = threadIdx.x;
  const int lane = tid & 31;
  const int wave = tid >> 5;

  // --- cooperative weight conversion into swizzled bf16 ---
  for (int idx = tid; idx < 32 * 64; idx += 256) {
    const int k = idx >> 6, n = idx & 63;
    W0s[bsw_index(k, n, 4)] =
        (bf16_t)((k < 8) ? w0[k * 64 + n] * 0.3535533905932738f : 0.0f);
  }
  for (int idx = tid; idx < 64 * 64; idx += 256) {
    const int k = idx >> 6, n = idx & 63;
    W1s[bsw_index(k, n, 4)] = (bf16_t)(w1[idx] * 0.125f);
    W2s[bsw_index(k, n, 4)] = (bf16_t)(w2[idx] * 0.125f);
  }
  for (int idx = tid; idx < 64 * 512; idx += 256) {
    const int k = idx >> 9, n = idx & 511;
    W3s[bsw_index(k, n, 32)] = (bf16_t)(w3[idx] * 0.125f);
  }

  // --- per-wave edge metadata ---
  const int eb = blockIdx.x * 128 + wave * 16;
  for (int t = lane; t < 16; t += 32) {
    const int e  = eb + t;
    const int ec = (e < N_EDGES) ? e : (N_EDGES - 1);
    eis[wave * 16 + t] = edge_i[ec];
    ejs[wave * 16 + t] = edge_j[ec];
  }
  for (int t = lane; t < 256; t += 32) {
    const int e  = eb + (t >> 4);
    const int ec = (e < N_EDGES) ? e : (N_EDGES - 1);
    sphs[wave * 256 + t] = sph[(size_t)ec * 16 + (t & 15)];
  }
  __syncthreads();

  bf16_t* stg = stage + wave * 1024;    // 2 ktiles of A-fragments

  // --- A0: radial basis tile (K padded 8->32).  Unconditional clamped b128
  // loads + per-element cndmask select (no branches). ---
  v16bf a0;
  {
    const int  m  = lane & 15;
    const int  e  = eb + m;
    const int  ec = (e < N_EDGES) ? e : (N_EDGES - 1);
    const bool lo = (lane < 16);       // only lane half 0 carries K=0..7
    float t[8];
    const float* p = rb + (size_t)ec * 8;
    *(float4*)(t + 0) = ld4(p);
    *(float4*)(t + 4) = ld4(p + 4);
#pragma unroll
    for (int j = 0; j < 16; ++j)
      a0[j] = (j < 8 && lo) ? (bf16_t)t[j] : (bf16_t)0.0f;
  }

  // --- layer 0: (16x32)@(32x64) ---
  {
    v8f h[4];
#pragma unroll
    for (int nt = 0; nt < 4; ++nt)
      h[nt] = wmma_bf16(a0, load_frag_lds(W0s, nt, lane), zero_v8f());
    store_silu(stg, h, lane);
  }
  // --- layer 1: (16x64)@(64x64) --- (reads precede writes; same-wave LDS
  // ops are in-order, so the single stage buffer is safe)
  {
    const v16bf alo = load_frag_lds(stg, 0, lane);
    const v16bf ahi = load_frag_lds(stg, 1, lane);
    v8f h[4];
#pragma unroll
    for (int nt = 0; nt < 4; ++nt) {
      v8f c = wmma_bf16(alo, load_frag_lds(W1s, 0 * 4 + nt, lane), zero_v8f());
      h[nt] = wmma_bf16(ahi, load_frag_lds(W1s, 1 * 4 + nt, lane), c);
    }
    store_silu(stg, h, lane);
  }
  // --- layer 2 ---
  {
    const v16bf alo = load_frag_lds(stg, 0, lane);
    const v16bf ahi = load_frag_lds(stg, 1, lane);
    v8f h[4];
#pragma unroll
    for (int nt = 0; nt < 4; ++nt) {
      v8f c = wmma_bf16(alo, load_frag_lds(W2s, 0 * 4 + nt, lane), zero_v8f());
      h[nt] = wmma_bf16(ahi, load_frag_lds(W2s, 1 * 4 + nt, lane), c);
    }
    store_silu(stg, h, lane);
  }

  // --- final layer (16x64)@(64x512) tile-by-tile, fused with scatter ---
  const v16bf alo = load_frag_lds(stg, 0, lane);
  const v16bf ahi = load_frag_lds(stg, 1, lane);

  const int mh = (lane >> 4) << 3;
  int  ei_r[8], ej_r[8];
  bool ev[8];
#pragma unroll
  for (int r = 0; r < 8; ++r) {
    const int m = r + mh;
    ei_r[r] = eis[wave * 16 + m];
    ej_r[r] = ejs[wave * 16 + m];
    ev[r]   = (eb + m) < N_EDGES;
  }

  for (int nt = 0; nt < 32; ++nt) {
    v8f c = wmma_bf16(alo, load_frag_lds(W3s, 0 * 32 + nt, lane), zero_v8f());
    c     = wmma_bf16(ahi, load_frag_lds(W3s, 1 * 32 + nt, lane), c);

    const int l   = nt >> 3;                       // 8 N-tiles per l
    const int off = (l == 0) ? 0 : (l == 1) ? 1 : (l == 2) ? 4 : 9;
    const int d   = 2 * l + 1;
    const int u   = ((nt & 7) << 4) + (lane & 15); // channel within MUL

#pragma unroll
    for (int r = 0; r < 8; ++r) {
      if (!ev[r]) continue;
      const int   m    = r + mh;
      const float prod = c[r] * x[(size_t)ej_r[r] * 128 + u];
      float* dst = tot + (size_t)off * (N_NODES * 128) + (size_t)ei_r[r] * 128 + u;
      const float* ys = sphs + wave * 256 + m * 16 + off;
      for (int k = 0; k < d; ++k)
        atomic_add_relaxed(dst + (size_t)k * (N_NODES * 128), ys[k] * prod);
    }
  }
}

// ---------------------------------------------------------------------------
extern "C" void kernel_launch(void* const* d_in, const int* in_sizes, int n_in,
                              void* d_out, int out_size, void* d_ws, size_t ws_size,
                              hipStream_t stream) {
  const float* node_features = (const float*)d_in[0];
  const float* sph           = (const float*)d_in[1];
  const float* rb            = (const float*)d_in[2];
  const int*   edge_i        = (const int*)d_in[3];
  const int*   edge_j        = (const int*)d_in[4];
  const float* W_pre         = (const float*)d_in[5];
  const float* w0            = (const float*)d_in[6];
  const float* w1            = (const float*)d_in[7];
  const float* w2            = (const float*)d_in[8];
  const float* w3            = (const float*)d_in[9];
  const float* W_post        = (const float*)d_in[10];

  float* out = (float*)d_out;
  float* x   = (float*)d_ws;                 // 5000*128 f32
  float* tot = x + (size_t)N_NODES * 128;    // 16*5000*128 f32, channel-major

  zero_kernel<<<2048, 256, 0, stream>>>(tot, 16 * N_NODES * 128);

  gemm_pre_kernel<<<dim3((N_NODES + 15) / 16), 256, 0, stream>>>(node_features, W_pre, x);

  const size_t smem_bytes =
      (size_t)(32 * 64 + 64 * 64 + 64 * 64 + 64 * 512) * sizeof(bf16_t)  // 86016
      + (size_t)(8 * 1024) * sizeof(bf16_t)                              // 16384
      + (size_t)(8 * 256) * sizeof(float)                                // 8192
      + (size_t)256 * sizeof(int);                                       // 1024
  hipFuncSetAttribute((const void*)edge_kernel,
                      hipFuncAttributeMaxDynamicSharedMemorySize,
                      (int)smem_bytes);
  edge_kernel<<<dim3((N_EDGES + 127) / 128), 256, smem_bytes, stream>>>(
      rb, sph, edge_i, edge_j, x, w0, w1, w2, w3, tot);

  gemm_post_kernel<<<dim3((N_NODES + 15) / 16, 16), 256, 0, stream>>>(tot, W_post, out);
}
